// VoxelBackBone8xFuse_89017492177631
// MI455X (gfx1250) — matrix operand order repaired and checked by reading, not compiled, not run on hardware
//
#include <hip/hip_runtime.h>

// ---------------------------------------------------------------------------
// VoxelBackBone8xFuse for MI455X (gfx1250): channel-last bf16 implicit-GEMM
// conv3d layers driven by v_wmma_f32_16x16x32_bf16 (wave32 WMMA), fused
// BN+ReLU+mask epilogue. Memory-bound problem (~550MB traffic, ~86 GFLOP),
// so bf16 storage + WMMA accumulate-in-f32 is the chip-rewarded path.
// Boundary taps read a zeroed guard page instead of predicated loads, so the
// inner loop is pure b128 loads + wmma (no exec-mask branching).
// ---------------------------------------------------------------------------

typedef unsigned short u16_t;
typedef unsigned int   u32_t;
typedef __attribute__((ext_vector_type(16))) __bf16 v16bf;
typedef __attribute__((ext_vector_type(8)))  float  v8f;
typedef __attribute__((ext_vector_type(4)))  u32_t  u32x4;

__device__ __forceinline__ u16_t f2bf(float f) {
  union { float f; u32_t u; } x; x.f = f;
  u32_t u = x.u;
  return (u16_t)((u + 0x7FFFu + ((u >> 16) & 1u)) >> 16);   // RNE
}
__device__ __forceinline__ float bf2f(u16_t h) {
  union { u32_t u; float f; } x; x.u = ((u32_t)h) << 16; return x.f;
}

// ---------------------------------------------------------------------------
// Generic implicit-GEMM conv3d. Activations: bf16, channel-last, C padded to
// CinPad (multiple of 32). Weights pre-packed in WMMA B-operand lane order:
// wp[((t*ksTot + ks)*nNT + nt)*512 + lane*16 + e], with
//   K = ks*32 + (e>>3)*16 + (lane>>4)*8 + (e&7),   N = nt*16 + (lane&15).
// A-operand per ISA 16-bit A 16x32 table: lane holds row M=lane&15; VGPR p
// holds K pair = ks*32 + (p<4?0:16) + (p&3)*2 + (lane>>4)*8. Those pairs are
// contiguous dwords -> two 16B quad loads per k-step (offset 0 and +32B).
// One wave = 16 output voxels x 16 output channels.
// ---------------------------------------------------------------------------
__global__ __launch_bounds__(256) void conv3d_wmma_bf16(
    const u16_t* __restrict__ in, const u16_t* __restrict__ zg,
    const u16_t* __restrict__ wp,
    const float* __restrict__ bnS, const float* __restrict__ bnT,
    const unsigned char* __restrict__ omask,
    u16_t* __restrict__ outBf, float* __restrict__ outF32, int dense,
    int Din, int Hin, int Win, int CinPad,
    int Dout, int Hout, int Wout, int CoPad,
    int KD, int KH, int KW, int sd, int sh, int sw, int pd, int ph, int pw)
{
  const int lane = threadIdx.x & 31;
  const int wave = blockIdx.x * (blockDim.x >> 5) + (threadIdx.x >> 5);
  const int Vout = Dout * Hout * Wout;
  const int nNT  = CoPad >> 4;
  const int nVT  = (Vout + 15) >> 4;
  if (wave >= nVT * nNT) return;          // wave-uniform exit: EXEC stays full

  const int vt = wave / nNT;
  const int nt = wave % nNT;
  const int m  = lane & 15;               // A-operand row this lane feeds
  const int hf = lane >> 4;               // half-wave selector
  const int v  = vt * 16 + m;
  const bool vok = (v < Vout);
  const int vc = vok ? v : 0;
  const int ow = vc % Wout;
  const int tq = vc / Wout;
  const int oh = tq % Hout;
  const int od = tq / Hout;
  const int ksTot  = CinPad >> 5;
  const int nBstep = nNT * 512;           // packed-weight stride per (t,ks)

  const u16_t* wr = wp + nt * 512 + lane * 16;   // walks packed order linearly
  v8f acc = {0.f, 0.f, 0.f, 0.f, 0.f, 0.f, 0.f, 0.f};

  for (int kd0 = 0; kd0 < KD; ++kd0) {
    const int  id   = od * sd - pd + kd0;
    const bool idok = vok && ((unsigned)id < (unsigned)Din);
    for (int kh0 = 0; kh0 < KH; ++kh0) {
      const int  ih   = oh * sh - ph + kh0;
      const bool ihok = idok && ((unsigned)ih < (unsigned)Hin);
      const long long rowbase = ((long long)id * Hin + ih) * Win;
      for (int kw0 = 0; kw0 < KW; ++kw0) {
        const int  iw  = ow * sw - pw + kw0;
        const bool inb = ihok && ((unsigned)iw < (unsigned)Win);
        // Pointer select: OOB taps read the zero guard page -> unconditional
        // loads, no exec-mask juggling in the hot loop.
        const u16_t* src = inb ? (in + (rowbase + iw) * (long long)CinPad) : zg;
        for (int ks = 0; ks < ksTot; ++ks) {
          union { u32x4 q[2]; v16bf b; } A;
          const u32x4* aq = (const u32x4*)(src + ks * 32 + hf * 8);
          A.q[0] = aq[0];      // K pairs {0,2,4,6}  (+hf*8, +ks*32)
          A.q[1] = aq[2];      // K pairs {16,18,20,22} : +32 bytes
          union { u32x4 q[2]; v16bf b; } B;
          const u32x4* bq = (const u32x4*)wr;
          B.q[0] = bq[0];
          B.q[1] = bq[1];
          acc = __builtin_amdgcn_wmma_f32_16x16x32_bf16(
              false, A.b, false, B.b, (short)0, acc, false, false);
          wr += nBstep;
        }
      }
    }
  }

  // Epilogue: D layout (f32 16x16): lane 0-15 -> N=lane, VGPR r -> M=r;
  // lane 16-31 -> N=lane-16, VGPR r -> M=r+8.  Fuse BN affine + ReLU + mask.
  const int n  = lane & 15;
  const int co = nt * 16 + n;
  const float sC = bnS[co];
  const float tC = bnT[co];
#pragma unroll
  for (int r = 0; r < 8; ++r) {
    const int mrow = r + hf * 8;
    const int vox  = vt * 16 + mrow;
    if (vox < Vout) {
      float val = acc[r] * sC + tC;
      val = val > 0.f ? val : 0.f;
      val *= (float)omask[vox];
      if (dense) outF32[(long long)co * Vout + vox] = val;     // NCDHW f32
      else       outBf[(long long)vox * CoPad + co] = f2bf(val);
    }
  }
}

// ---------------------------------------------------------------------------
// Weight packing: fp32 (Co,Cin,KD,KH,KW) -> bf16 WMMA-B lane order, zero-padded
// to (CinPad, CoPad).
// ---------------------------------------------------------------------------
__global__ void pack_w_k(const float* __restrict__ w, u16_t* __restrict__ wp,
                         int Cin, int Co, int CinPad, int CoPad,
                         int KD, int KH, int KW)
{
  const int taps = KD * KH * KW;
  const int ksT  = CinPad >> 5;
  const int nNT  = CoPad >> 4;
  const int total = taps * ksT * nNT * 512;
  int idx = blockIdx.x * blockDim.x + threadIdx.x;
  if (idx >= total) return;
  const int e    = idx & 15;
  const int lane = (idx >> 4) & 31;
  int rest = idx >> 9;
  const int nt = rest % nNT; rest /= nNT;
  const int ks = rest % ksT;
  const int t  = rest / ksT;
  const int k = ks * 32 + (e >> 3) * 16 + (lane >> 4) * 8 + (e & 7);
  const int n = nt * 16 + (lane & 15);
  float val = 0.f;
  if (k < Cin && n < Co) {
    const int kw0 = t % KW;
    const int kh0 = (t / KW) % KH;
    const int kd0 = t / (KW * KH);
    val = w[(((long long)n * Cin + k) * KD + kd0) * (KH * KW) + kh0 * KW + kw0];
  }
  wp[idx] = f2bf(val);
}

// BN fold: s = g/sqrt(v+eps), t = b - m*s ; zero beyond Co so padded channels
// stay exactly zero through the ReLU epilogue.
__global__ void pack_bn_k(const float* __restrict__ g, const float* __restrict__ b,
                          const float* __restrict__ mu, const float* __restrict__ vv,
                          float* __restrict__ s, float* __restrict__ t,
                          int Co, int CoPad)
{
  int c = blockIdx.x * blockDim.x + threadIdx.x;
  if (c >= CoPad) return;
  if (c < Co) {
    float sc = g[c] * rsqrtf(vv[c] + 1e-3f);
    s[c] = sc;
    t[c] = b[c] - mu[c] * sc;
  } else {
    s[c] = 0.f;
    t[c] = 0.f;
  }
}

// Zero guard page (256B) read by out-of-bounds conv taps.
__global__ void zguard_k(u32_t* __restrict__ g)
{
  g[threadIdx.x] = 0u;   // <<<1,64>>> -> 256 bytes
}

// Input prep: NCDHW fp32 x * mask -> channel-last bf16 padded to 32 channels.
__global__ void prep_k(const float* __restrict__ x, const unsigned char* __restrict__ mask,
                       u16_t* __restrict__ out, int V)
{
  int v = blockIdx.x * blockDim.x + threadIdx.x;
  if (v >= V) return;
  const float mm = mask[v] ? 1.f : 0.f;
  union { u16_t h[32]; u32x4 q[4]; } row;
#pragma unroll
  for (int c = 0; c < 32; ++c) {
    float val = (c < 4) ? x[(long long)c * V + v] * mm : 0.f;
    row.h[c] = f2bf(val);
  }
  u32x4* dst = (u32x4*)(out + (long long)v * 32);
#pragma unroll
  for (int qi = 0; qi < 4; ++qi) dst[qi] = row.q[qi];
}

// Active-site propagation for strided sparse conv: OR over receptive field.
__global__ void mask_down_k(const unsigned char* __restrict__ mi, unsigned char* __restrict__ mo,
                            int Din, int Hin, int Win, int Dout, int Hout, int Wout,
                            int KD, int KH, int KW, int sd, int sh, int sw,
                            int pd, int ph, int pw)
{
  int v = blockIdx.x * blockDim.x + threadIdx.x;
  const int V = Dout * Hout * Wout;
  if (v >= V) return;
  const int ow = v % Wout;
  const int tq = v / Wout;
  const int oh = tq % Hout;
  const int od = tq / Hout;
  unsigned char a = 0;
  for (int kd0 = 0; kd0 < KD; ++kd0) {
    const int id = od * sd - pd + kd0;
    if (id < 0 || id >= Din) continue;
    for (int kh0 = 0; kh0 < KH; ++kh0) {
      const int ih = oh * sh - ph + kh0;
      if (ih < 0 || ih >= Hin) continue;
      for (int kw0 = 0; kw0 < KW; ++kw0) {
        const int iw = ow * sw - pw + kw0;
        if (iw < 0 || iw >= Win) continue;
        a |= (mi[(id * Hin + ih) * Win + iw] ? 1 : 0);
      }
    }
  }
  mo[v] = a;
}

// Channel-gate fuse at x_conv4: x *= (1 + sigmoid(fg*w[c] + b[c])).
__global__ void fuse_k(u16_t* __restrict__ x4, const float* __restrict__ fg,
                       const float* __restrict__ mw, const float* __restrict__ mb, int V)
{
  int idx = blockIdx.x * blockDim.x + threadIdx.x;
  if (idx >= V * 64) return;
  const int v = idx >> 6;
  const int c = idx & 63;
  const float z = fg[v] * mw[c] + mb[c];
  const float lw = 1.f / (1.f + __expf(-z));
  const long long o = (long long)v * 64 + c;
  x4[o] = f2bf(bf2f(x4[o]) * (1.f + lw));
}

// ---------------------------------------------------------------------------
// Host orchestration
// ---------------------------------------------------------------------------
struct Spec {
  int Cin, Co, CinPad, CoPad;
  int KD, KH, KW;
  int Din, Hin, Win, Dout, Hout, Wout;
  int sd, sh, sw, pd, ph, pw;
};

extern "C" void kernel_launch(void* const* d_in, const int* in_sizes, int n_in,
                              void* d_out, int out_size, void* d_ws, size_t ws_size,
                              hipStream_t stream)
{
  (void)in_sizes; (void)n_in; (void)out_size; (void)ws_size;

  const float*         x     = (const float*)d_in[0];
  const unsigned char* mask1 = (const unsigned char*)d_in[1];   // bool8
  const float*         fg    = (const float*)d_in[2];
  const float*         mlp_w = (const float*)d_in[63];
  const float*         mlp_b = (const float*)d_in[64];

  static const Spec LS[12] = {
    {  4,  16, 32, 32, 3,3,3, 41,176,200, 41,176,200, 1,1,1, 1,1,1 }, // in
    { 16,  16, 32, 32, 3,3,3, 41,176,200, 41,176,200, 1,1,1, 1,1,1 }, // c1
    { 16,  32, 32, 32, 3,3,3, 41,176,200, 21, 88,100, 2,2,2, 1,1,1 }, // c2a
    { 32,  32, 32, 32, 3,3,3, 21, 88,100, 21, 88,100, 1,1,1, 1,1,1 }, // c2b
    { 32,  32, 32, 32, 3,3,3, 21, 88,100, 21, 88,100, 1,1,1, 1,1,1 }, // c2c
    { 32,  64, 32, 64, 3,3,3, 21, 88,100, 11, 44, 50, 2,2,2, 1,1,1 }, // c3a
    { 64,  64, 64, 64, 3,3,3, 11, 44, 50, 11, 44, 50, 1,1,1, 1,1,1 }, // c3b
    { 64,  64, 64, 64, 3,3,3, 11, 44, 50, 11, 44, 50, 1,1,1, 1,1,1 }, // c3c
    { 64,  64, 64, 64, 3,3,3, 11, 44, 50,  5, 22, 25, 2,2,2, 0,1,1 }, // c4a
    { 64,  64, 64, 64, 3,3,3,  5, 22, 25,  5, 22, 25, 1,1,1, 1,1,1 }, // c4b
    { 64,  64, 64, 64, 3,3,3,  5, 22, 25,  5, 22, 25, 1,1,1, 1,1,1 }, // c4c
    { 64, 128, 64,128, 3,1,1,  5, 22, 25,  2, 22, 25, 2,1,1, 0,0,0 }, // out
  };

  const long long V1 = 41LL * 176 * 200;
  const long long V2 = 21LL * 88 * 100;
  const long long V3 = 11LL * 44 * 50;
  const long long V4 = 5LL * 22 * 25;
  const long long V5 = 2LL * 22 * 25;

  char* ws = (char*)d_ws;
  size_t off = 0;
  auto alloc = [&](size_t bytes) -> size_t {
    size_t o = off;
    off += (bytes + 255) & ~(size_t)255;
    return o;
  };

  const size_t buf0 = alloc((size_t)V1 * 32 * 2);   // res1-sized ping
  const size_t buf1 = alloc((size_t)V1 * 32 * 2);   // res1-sized pong
  const size_t zg   = alloc((size_t)256);           // zero guard page
  const size_t om2  = alloc((size_t)V2);
  const size_t om3  = alloc((size_t)V3);
  const size_t om4  = alloc((size_t)V4);
  const size_t om5  = alloc((size_t)V5);

  size_t wp_off[12], s_off[12], t_off[12];
  for (int i = 0; i < 12; ++i) {
    const Spec& L = LS[i];
    const int taps = L.KD * L.KH * L.KW;
    const int ksT  = L.CinPad >> 5;
    const int nNT  = L.CoPad >> 4;
    wp_off[i] = alloc((size_t)taps * ksT * nNT * 512 * 2);
    s_off[i]  = alloc((size_t)L.CoPad * 4);
    t_off[i]  = alloc((size_t)L.CoPad * 4);
  }

  // ---- zero guard + fold BN + pack weights (deterministic every call) ----
  zguard_k<<<1, 64, 0, stream>>>((u32_t*)(ws + zg));
  for (int i = 0; i < 12; ++i) {
    const Spec& L = LS[i];
    const float* w_  = (const float*)d_in[3 + 5 * i + 0];
    const float* g_  = (const float*)d_in[3 + 5 * i + 1];
    const float* b_  = (const float*)d_in[3 + 5 * i + 2];
    const float* m_  = (const float*)d_in[3 + 5 * i + 3];
    const float* v_  = (const float*)d_in[3 + 5 * i + 4];
    pack_bn_k<<<(L.CoPad + 63) / 64, 64, 0, stream>>>(
        g_, b_, m_, v_, (float*)(ws + s_off[i]), (float*)(ws + t_off[i]),
        L.Co, L.CoPad);
    const int total = L.KD * L.KH * L.KW * (L.CinPad >> 5) * (L.CoPad >> 4) * 512;
    pack_w_k<<<(total + 255) / 256, 256, 0, stream>>>(
        w_, (u16_t*)(ws + wp_off[i]),
        L.Cin, L.Co, L.CinPad, L.CoPad, L.KD, L.KH, L.KW);
  }

  // ---- input prep: x * mask -> bf16 channel-last (pad to 32) ----
  prep_k<<<(int)((V1 + 127) / 128), 128, 0, stream>>>(x, mask1, (u16_t*)(ws + buf0), (int)V1);

  // ---- active-site mask propagation chain ----
  mask_down_k<<<(int)((V2 + 127) / 128), 128, 0, stream>>>(
      mask1, (unsigned char*)(ws + om2), 41,176,200, 21,88,100, 3,3,3, 2,2,2, 1,1,1);
  mask_down_k<<<(int)((V3 + 127) / 128), 128, 0, stream>>>(
      (unsigned char*)(ws + om2), (unsigned char*)(ws + om3),
      21,88,100, 11,44,50, 3,3,3, 2,2,2, 1,1,1);
  mask_down_k<<<(int)((V4 + 127) / 128), 128, 0, stream>>>(
      (unsigned char*)(ws + om3), (unsigned char*)(ws + om4),
      11,44,50, 5,22,25, 3,3,3, 2,2,2, 0,1,1);
  mask_down_k<<<(int)((V5 + 127) / 128), 128, 0, stream>>>(
      (unsigned char*)(ws + om4), (unsigned char*)(ws + om5),
      5,22,25, 2,22,25, 3,1,1, 2,1,1, 0,0,0);

  // ---- conv chain (ping-pong buf0 <-> buf1; final layer -> d_out f32) ----
  for (int i = 0; i < 12; ++i) {
    const Spec& L = LS[i];
    const u16_t* inP  = (const u16_t*)(ws + ((i & 1) ? buf1 : buf0));
    u16_t*       outP = (u16_t*)(ws + ((i & 1) ? buf0 : buf1));
    const unsigned char* omask =
        (i < 2)  ? mask1 :
        (i < 5)  ? (const unsigned char*)(ws + om2) :
        (i < 8)  ? (const unsigned char*)(ws + om3) :
        (i < 11) ? (const unsigned char*)(ws + om4)
                 : (const unsigned char*)(ws + om5);
    const int dense = (i == 11) ? 1 : 0;

    if (i == 11) {
      // Fuse channel gate at x_conv4 (input of conv_out lives in buf1).
      fuse_k<<<(int)((V4 * 64 + 255) / 256), 256, 0, stream>>>(
          (u16_t*)(ws + buf1), fg, mlp_w, mlp_b, (int)V4);
    }

    const int Vout = L.Dout * L.Hout * L.Wout;
    const int nWaves = ((Vout + 15) >> 4) * (L.CoPad >> 4);
    const int blocks = (nWaves + 7) / 8;   // 8 waves (256 threads) per block
    conv3d_wmma_bf16<<<blocks, 256, 0, stream>>>(
        inP, (const u16_t*)(ws + zg), (const u16_t*)(ws + wp_off[i]),
        (const float*)(ws + s_off[i]), (const float*)(ws + t_off[i]),
        omask, outP, (float*)d_out, dense,
        L.Din, L.Hin, L.Win, L.CinPad,
        L.Dout, L.Hout, L.Wout, L.CoPad,
        L.KD, L.KH, L.KW, L.sd, L.sh, L.sw, L.pd, L.ph, L.pw);
  }
}
